// VAE_65859028516878
// MI455X (gfx1250) — matrix operand order repaired and checked by reading, not compiled, not run on hardware
//
#include <hip/hip_runtime.h>
#include <math.h>

// ---------------------------------------------------------------------------
// VAE-ish pipeline for MI455X (gfx1250, wave32, WMMA).
//   GEMM1: relu([data|omega] @ W_feat1 + b)           -> f16 [N,256]
//   GEMM2: relu(feat1 @ W_feat2 + b)                  -> f16 [N,256]
//   GEMM3: relu(feat @ [W_xim1|W_xil1] + [b|b])       -> f16 [N,256]
//   heads: 128-dots + softplus / lnvar / xi=clip(...)
//   ODE:   256 sequential steps, 2->64->1 MLP, 4 lanes per sample,
//          weights in registers, shuffle-reduced; x_PAB staged time-major
//   transpose: [256][N] -> out [N][256]
// GEMMs use v_wmma_f32_16x16x32_f16 (f32 accumulate) with explicit A/B
// double-buffering so WMMAs consume tiles loaded one k-step earlier
// (breaks the load->wait0->wmma serialization seen in the round-1 asm).
// ---------------------------------------------------------------------------

#define NN   16384
#define DIM  256
#define DTF  0.01f

typedef __attribute__((ext_vector_type(16))) _Float16 v16h;
typedef __attribute__((ext_vector_type(8)))  float    v8f;
typedef __attribute__((ext_vector_type(4)))  float    f32x4;

union V16H { v16h v; f32x4 f[2]; };

static __device__ __forceinline__ v8f wmma16x16x32(v16h a, v16h b, v8f c) {
  // 8 args: (neg_a, A, neg_b, B, c_mod, C, reuse_a, reuse_b)
  return __builtin_amdgcn_wmma_f32_16x16x32_f16(false, a, false, b, (short)0, c,
                                                false, false);
}

static __device__ __forceinline__ void load_b_tile(const _Float16* p, V16H& b) {
  b.f[0] = *(const f32x4*)(p);       // K = kb+8h    .. +7
  b.f[1] = *(const f32x4*)(p + 16);  // K = kb+16+8h .. +7
}

// ---------------------------------------------------------------------------
// Weight prep: transpose f32 weights into f16 [n][k] tables (B-operand layout
// here loads two runs of 8 contiguous K values per lane).
// ---------------------------------------------------------------------------
__global__ void convert_weights(const float* __restrict__ Wf1,   // [257][256]
                                const float* __restrict__ Wf2,   // [256][256]
                                const float* __restrict__ Wxm1,  // [256][128]
                                const float* __restrict__ Wxl1,  // [256][128]
                                _Float16* __restrict__ wt1,
                                _Float16* __restrict__ wt2,
                                _Float16* __restrict__ wth) {
  const int k = threadIdx.x;       // 0..255
  const int j = blockIdx.x;        // 0..255 output column
  const int w = blockIdx.y;        // which matrix
  if (w == 0) {
    wt1[j * DIM + k] = (_Float16)Wf1[k * DIM + j];   // rows 0..255 only
  } else if (w == 1) {
    wt2[j * DIM + k] = (_Float16)Wf2[k * DIM + j];
  } else {
    wth[j * DIM + k] = (_Float16)(j < 128 ? Wxm1[k * 128 + j]
                                          : Wxl1[k * 128 + (j - 128)]);
  }
}

// ---------------------------------------------------------------------------
// GEMM1: A = data f32 [N][256] (converted to f16 on the fly), omega column
// (K=256 of the 257-wide input) folded into the accumulator init.
// Software-pipelined: raw A f32 quads + B tiles for kb+32 are in flight while
// the WMMAs for kb execute.
// ---------------------------------------------------------------------------
__global__ void __launch_bounds__(128)
gemm1_relu(const float* __restrict__ A, const float* __restrict__ omega,
           const _Float16* __restrict__ Wt, const float* __restrict__ bias,
           const float* __restrict__ wlast,   // W_feat1 row 256
           _Float16* __restrict__ Out) {
  const int lane = threadIdx.x & 31;
  const int wave = threadIdx.x >> 5;
  const int ml   = lane & 15;   // A row within tile / B,C column within tile
  const int kh   = lane >> 4;   // half selector
  const int rowBase = blockIdx.x * 16;

  float om[8];
#pragma unroll
  for (int r = 0; r < 8; ++r) om[r] = omega[rowBase + 8 * kh + r];

  v8f acc[4];
#pragma unroll
  for (int s = 0; s < 4; ++s) {
    const int n  = wave * 64 + s * 16 + ml;
    const float b  = bias[n];
    const float wl = wlast[n];
#pragma unroll
    for (int r = 0; r < 8; ++r) acc[s][r] = fmaf(om[r], wl, b);
  }

  const float*    arow = A  + (rowBase + ml) * DIM + 8 * kh;
  const _Float16* wcol = Wt + (wave * 64 + ml) * DIM + 8 * kh;

  // prologue: stage kb = 0
  f32x4 d0 = *(const f32x4*)(arow);
  f32x4 d1 = *(const f32x4*)(arow + 4);
  f32x4 d2 = *(const f32x4*)(arow + 16);
  f32x4 d3 = *(const f32x4*)(arow + 20);
  V16H b[4];
#pragma unroll
  for (int s = 0; s < 4; ++s) load_b_tile(wcol + s * 16 * DIM, b[s]);

  for (int kb = 0; kb < DIM; kb += 32) {
    const int kn = (kb + 32) & 255;        // wraps harmlessly on last iter
    // issue next-iteration loads first
    f32x4 e0 = *(const f32x4*)(arow + kn);
    f32x4 e1 = *(const f32x4*)(arow + kn + 4);
    f32x4 e2 = *(const f32x4*)(arow + kn + 16);
    f32x4 e3 = *(const f32x4*)(arow + kn + 20);
    V16H bn[4];
#pragma unroll
    for (int s = 0; s < 4; ++s) load_b_tile(wcol + s * 16 * DIM + kn, bn[s]);

    // convert staged f32 -> f16 A operand and compute
    V16H a;
#pragma unroll
    for (int i = 0; i < 4; ++i) {
      a.v[i]      = (_Float16)d0[i];
      a.v[4 + i]  = (_Float16)d1[i];
      a.v[8 + i]  = (_Float16)d2[i];
      a.v[12 + i] = (_Float16)d3[i];
    }
#pragma unroll
    for (int s = 0; s < 4; ++s) acc[s] = wmma16x16x32(a.v, b[s].v, acc[s]);

    d0 = e0; d1 = e1; d2 = e2; d3 = e3;
#pragma unroll
    for (int s = 0; s < 4; ++s) b[s] = bn[s];
  }

#pragma unroll
  for (int s = 0; s < 4; ++s) {
    const int n = wave * 64 + s * 16 + ml;
#pragma unroll
    for (int r = 0; r < 8; ++r) {
      float x = acc[s][r];
      x = x > 0.f ? x : 0.f;
      Out[(rowBase + 8 * kh + r) * DIM + n] = (_Float16)x;
    }
  }
}

// ---------------------------------------------------------------------------
// GEMM (f16 A) + bias (split across two vectors) + relu -> f16 out.
// Used for GEMM2 (bsplit=256) and GEMM3 (bsplit=128, concat heads).
// Same double-buffered pipeline as gemm1.
// ---------------------------------------------------------------------------
__global__ void __launch_bounds__(128)
gemm_f16_relu(const _Float16* __restrict__ A, const _Float16* __restrict__ Wt,
              const float* __restrict__ bias0, const float* __restrict__ bias1,
              int bsplit, _Float16* __restrict__ Out) {
  const int lane = threadIdx.x & 31;
  const int wave = threadIdx.x >> 5;
  const int ml   = lane & 15;
  const int kh   = lane >> 4;
  const int rowBase = blockIdx.x * 16;

  v8f acc[4];
#pragma unroll
  for (int s = 0; s < 4; ++s) {
    const int n = wave * 64 + s * 16 + ml;
    const float bb = (n < bsplit) ? bias0[n] : bias1[n - bsplit];
#pragma unroll
    for (int r = 0; r < 8; ++r) acc[s][r] = bb;
  }

  const _Float16* arow = A  + (rowBase + ml) * DIM + 8 * kh;
  const _Float16* wcol = Wt + (wave * 64 + ml) * DIM + 8 * kh;

  V16H a, b[4];
  a.f[0] = *(const f32x4*)(arow);
  a.f[1] = *(const f32x4*)(arow + 16);
#pragma unroll
  for (int s = 0; s < 4; ++s) load_b_tile(wcol + s * 16 * DIM, b[s]);

  for (int kb = 0; kb < DIM; kb += 32) {
    const int kn = (kb + 32) & 255;
    V16H an, bn[4];
    an.f[0] = *(const f32x4*)(arow + kn);
    an.f[1] = *(const f32x4*)(arow + kn + 16);
#pragma unroll
    for (int s = 0; s < 4; ++s) load_b_tile(wcol + s * 16 * DIM + kn, bn[s]);

#pragma unroll
    for (int s = 0; s < 4; ++s) acc[s] = wmma16x16x32(a.v, b[s].v, acc[s]);

    a = an;
#pragma unroll
    for (int s = 0; s < 4; ++s) b[s] = bn[s];
  }

#pragma unroll
  for (int s = 0; s < 4; ++s) {
    const int n = wave * 64 + s * 16 + ml;
#pragma unroll
    for (int r = 0; r < 8; ++r) {
      float x = acc[s][r];
      x = x > 0.f ? x : 0.f;
      Out[(rowBase + 8 * kh + r) * DIM + n] = (_Float16)x;
    }
  }
}

// ---------------------------------------------------------------------------
// Head finalization: one wave32 per sample. 128-length dots via lane-parallel
// FMA + __shfl_xor reduction; softplus/exp/clip on lane 0.
// ---------------------------------------------------------------------------
__global__ void __launch_bounds__(256)
heads_final(const _Float16* __restrict__ H,    // [N][256], cols 0-127 mean head
            const float* __restrict__ Wm2, const float* __restrict__ bm2,
            const float* __restrict__ Wl2, const float* __restrict__ bl2,
            const float* __restrict__ eps,
            float* __restrict__ out_mean, float* __restrict__ out_lnvar,
            float* __restrict__ xi) {
  const int lane = threadIdx.x & 31;
  const int wave = threadIdx.x >> 5;
  const int s = blockIdx.x * 8 + wave;
  const _Float16* hr = H + s * DIM;

  float pm = 0.f, pl = 0.f;
#pragma unroll
  for (int i = 0; i < 4; ++i) {
    const int k = lane + 32 * i;
    pm = fmaf((float)hr[k],       Wm2[k], pm);
    pl = fmaf((float)hr[128 + k], Wl2[k], pl);
  }
#pragma unroll
  for (int off = 16; off > 0; off >>= 1) {
    pm += __shfl_xor(pm, off, 32);
    pl += __shfl_xor(pl, off, 32);
  }
  if (lane == 0) {
    const float zm = pm + bm2[0];
    const float xm = (zm > 0.f) ? zm + log1pf(expf(-zm)) : log1pf(expf(zm));
    const float zl = pl + bl2[0];
    float x = xm + expf(0.5f * zl) * eps[s];
    x = fminf(fmaxf(x, 0.f), 1.f);
    out_mean[s]  = xm;
    out_lnvar[s] = zl;
    xi[s]        = x;
  }
}

// ---------------------------------------------------------------------------
// ODE: 4 lanes per sample (lanes g&3 own hidden units u*16..u*16+15), weights
// and the omega-folded bias entirely in registers, 4-way split accumulator
// (chain depth 4), 2-shuffle cross-lane reduction per step. 65536 threads =
// 2048 wave32 (4x the fill of one-thread-per-sample, which left most SIMDs
// empty on a chip this wide). x_PAB written time-major (coalesced) by the
// u==0 lane for the later transpose.
// ---------------------------------------------------------------------------
__global__ void __launch_bounds__(256)
ode_kernel(const float* __restrict__ data, const float* __restrict__ omega,
           const float* __restrict__ xi_,
           const float* __restrict__ Wa1, const float* __restrict__ ba1,
           const float* __restrict__ Wa2, const float* __restrict__ ba2,
           float* __restrict__ xpab_t /* [256][NN] */) {
  const int g = blockIdx.x * 256 + threadIdx.x;
  const int s = g >> 2;        // sample
  const int u = g & 3;         // subunit: hidden units j = u*16 + i

  const float om = omega[s];
  float w11[16], cc[16], w2[16];
#pragma unroll
  for (int i = 0; i < 16; ++i) {
    const int j = u * 16 + i;
    w11[i] = Wa1[64 + j];                 // W_aux1[1][j]
    cc[i]  = fmaf(om, Wa1[j], ba1[j]);    // omega*W_aux1[0][j] + b_aux1[j]
    w2[i]  = Wa2[j];
  }
  const float axi = fabsf(xi_[s]);
  const float b2  = ba2[0];
  float y = data[s * DIM];     // init_y = data[:, 0]
  float v = 0.f;

  for (int t = 0; t < 256; ++t) {
    if (u == 0)
      xpab_t[t * NN + s] = expf(-axi * (DTF * (float)t)) * y;

    float a0 = 0.f, a1 = 0.f, a2 = 0.f, a3 = 0.f;
#pragma unroll
    for (int i = 0; i < 16; i += 4) {
      float h0 = fmaxf(fmaf(y, w11[i],     cc[i]),     0.f);
      float h1 = fmaxf(fmaf(y, w11[i + 1], cc[i + 1]), 0.f);
      float h2 = fmaxf(fmaf(y, w11[i + 2], cc[i + 2]), 0.f);
      float h3 = fmaxf(fmaf(y, w11[i + 3], cc[i + 3]), 0.f);
      a0 = fmaf(h0, w2[i],     a0);
      a1 = fmaf(h1, w2[i + 1], a1);
      a2 = fmaf(h2, w2[i + 2], a2);
      a3 = fmaf(h3, w2[i + 3], a3);
    }
    float acc = (a0 + a1) + (a2 + a3);
    acc += __shfl_xor(acc, 1, 32);     // combine the 4 lanes of this sample
    acc += __shfl_xor(acc, 2, 32);
    acc += b2;

    const float yn = fmaf(DTF, v, y);  // y += DT * v_old
    v = fmaf(DTF, acc, v);             // v += DT * acc
    y = yn;
  }
}

// ---------------------------------------------------------------------------
// Transpose [256][NN] -> out [NN][256] via 32x33 LDS tile (coalesced both
// directions, avoids 1 KB-stride scatter stores out of the ODE kernel).
// ---------------------------------------------------------------------------
__global__ void __launch_bounds__(256)
transpose_xpab(const float* __restrict__ src, float* __restrict__ dst) {
  __shared__ float tile[32][33];
  const int nx = blockIdx.x * 32 + threadIdx.x;   // sample index
  const int t0 = blockIdx.y * 32;
#pragma unroll
  for (int i = 0; i < 32; i += 8)
    tile[threadIdx.y + i][threadIdx.x] = src[(t0 + threadIdx.y + i) * NN + nx];
  __syncthreads();
  const int tx = blockIdx.y * 32 + threadIdx.x;   // time index
  const int n0 = blockIdx.x * 32;
#pragma unroll
  for (int i = 0; i < 32; i += 8)
    dst[(n0 + threadIdx.y + i) * 256 + tx] = tile[threadIdx.x][threadIdx.y + i];
}

// ---------------------------------------------------------------------------
// Launcher. Workspace layout (bytes):
//   [0,128K)    wt1 f16      [128K,256K) wt2 f16    [256K,384K) wth f16
//   [384K,448K) xi f32[N]
//   [512K, 512K+16M):
//     phase 1: bufA f16 [N][256] @ +0, bufB f16 [N][256] @ +8M
//              (GEMM1->bufA, GEMM2 bufA->bufB, GEMM3 bufB->bufA)
//     phase 2: xpab_t f32 [256][N] spans the full 16M (f16 bufs dead by then)
// Total required: ~16.5 MB.
// ---------------------------------------------------------------------------
extern "C" void kernel_launch(void* const* d_in, const int* in_sizes, int n_in,
                              void* d_out, int out_size, void* d_ws,
                              size_t ws_size, hipStream_t stream) {
  const float* data   = (const float*)d_in[0];
  const float* omega  = (const float*)d_in[1];
  const float* eps    = (const float*)d_in[2];
  const float* Wf1    = (const float*)d_in[3];
  const float* bf1    = (const float*)d_in[4];
  const float* Wf2    = (const float*)d_in[5];
  const float* bf2    = (const float*)d_in[6];
  const float* Wxm1   = (const float*)d_in[7];
  const float* bxm1   = (const float*)d_in[8];
  const float* Wxm2   = (const float*)d_in[9];
  const float* bxm2   = (const float*)d_in[10];
  const float* Wxl1   = (const float*)d_in[11];
  const float* bxl1   = (const float*)d_in[12];
  const float* Wxl2   = (const float*)d_in[13];
  const float* bxl2   = (const float*)d_in[14];
  const float* Wa1    = (const float*)d_in[15];
  const float* ba1    = (const float*)d_in[16];
  const float* Wa2    = (const float*)d_in[17];
  const float* ba2    = (const float*)d_in[18];

  char* ws = (char*)d_ws;
  const size_t KB = 1024, MB = 1024 * 1024;
  _Float16* wt1 = (_Float16*)(ws + 0);
  _Float16* wt2 = (_Float16*)(ws + 128 * KB);
  _Float16* wth = (_Float16*)(ws + 256 * KB);
  float*    xi  = (float*)(ws + 384 * KB);
  _Float16* bufA = (_Float16*)(ws + 512 * KB);
  _Float16* bufB = (_Float16*)(ws + 512 * KB + 8 * MB);
  float*    xpab = (float*)(ws + 512 * KB);

  float* out_mean  = (float*)d_out;
  float* out_lnvar = out_mean + NN;
  float* out_xpab  = out_mean + 2 * NN;

  convert_weights<<<dim3(256, 3), 256, 0, stream>>>(Wf1, Wf2, Wxm1, Wxl1,
                                                    wt1, wt2, wth);
  gemm1_relu<<<NN / 16, 128, 0, stream>>>(data, omega, wt1, bf1,
                                          Wf1 + 256 * 256 /* omega row */,
                                          bufA);
  gemm_f16_relu<<<NN / 16, 128, 0, stream>>>(bufA, wt2, bf2, bf2, 256, bufB);
  gemm_f16_relu<<<NN / 16, 128, 0, stream>>>(bufB, wth, bxm1, bxl1, 128, bufA);
  heads_final<<<NN / 8, 256, 0, stream>>>(bufA, Wxm2, bxm2, Wxl2, bxl2, eps,
                                          out_mean, out_lnvar, xi);
  ode_kernel<<<4 * NN / 256, 256, 0, stream>>>(data, omega, xi, Wa1, ba1, Wa2,
                                               ba2, xpab);
  transpose_xpab<<<dim3(NN / 32, 256 / 32), dim3(32, 8), 0, stream>>>(xpab,
                                                                      out_xpab);
}
